// Mamba2_64896955843200
// MI455X (gfx1250) — compile-verified, matrix operand order checked by reading, not run on hardware
//
#include <hip/hip_runtime.h>
#include <hip/hip_bf16.h>
#include <math.h>

// ---- problem constants (from reference) ----
#define D_MODEL   768
#define D_STATE   128
#define HEADDIM   64
#define D_INNER   1536                    // EXPAND * D_MODEL
#define NHEADS    24                      // D_INNER / HEADDIM
#define CONV_DIM  1792                    // D_INNER + 2*D_STATE
#define D_IN_PROJ 3352                    // 2*D_INNER + 2*D_STATE + NHEADS
#define BATCH     2
#define SEQLEN    512
#define NTOK      (BATCH * SEQLEN)        // 1024

typedef __attribute__((ext_vector_type(16))) _Float16 v16h;
typedef __attribute__((ext_vector_type(8)))  _Float16 v8h;
typedef __attribute__((ext_vector_type(8)))  float    v8f;

// =====================================================================
// fp32 -> f16 preconversion (memory-bound, done once per tensor).
// n must be a multiple of 4 (true for all tensors here).
// =====================================================================
__global__ __launch_bounds__(256)
void f32_to_f16_kernel(const float* __restrict__ in, _Float16* __restrict__ out,
                       int n4) {
    const int i = blockIdx.x * blockDim.x + threadIdx.x;
    if (i < n4) {
        const float4 v = ((const float4*)in)[i];
        v8h h;  // only 4 used; store as 8-byte chunk
        _Float16* o = out + i * 4;
        o[0] = (_Float16)v.x; o[1] = (_Float16)v.y;
        o[2] = (_Float16)v.z; o[3] = (_Float16)v.w;
    }
}

// Load one lane's 16-element fragment slice from f16 memory.
// Per CDNA5 ISA 7.12.2 (16-bit frag, 16x32): lane = (k/8 % 2)*16 + row,
// element e -> k = kb + e + 8*(e>=8), kb = k0 + 8*khalf.
// Two contiguous 8-half runs -> two aligned 16-byte loads.
__device__ __forceinline__ v16h load_frag_h(const _Float16* __restrict__ p) {
    const v8h lo = *(const v8h*)(p);        // k = kb + 0..7
    const v8h hi = *(const v8h*)(p + 16);   // k = kb + 16..23
    v16h r;
#pragma unroll
    for (int i = 0; i < 8; ++i) { r[i] = lo[i]; r[i + 8] = hi[i]; }
    return r;
}

// =====================================================================
// WMMA GEMM:  C[M x N] = A[M x K] * W[N x K]^T  (f16 in, fp32 out)
// 32x64 tile per wave (MSUB=2 x NSUB=4): 8 v_wmma per K-step fed by
// 12 x 16B loads (B fragments reused across both M-subtiles).
// Out-of-range N columns use a clamped row pointer and are not stored.
// =====================================================================
#define MSUB 2
#define NSUB 4

__global__ __launch_bounds__(128)
void wmma_gemm_nt(const _Float16* __restrict__ A, const _Float16* __restrict__ W,
                  float* __restrict__ C, int M, int N, int K) {
    const int lane = threadIdx.x & 31;
    const int wave = threadIdx.x >> 5;
    const int wavesPerBlock = blockDim.x >> 5;

    const int mgroups = M / (16 * MSUB);
    const int ngroups = (N + (16 * NSUB) - 1) / (16 * NSUB);
    const int tile    = blockIdx.x * wavesPerBlock + wave;
    if (tile >= mgroups * ngroups) return;         // wave-uniform branch

    const int mg = tile / ngroups;
    const int ng = tile % ngroups;

    const int mn    = lane & 15;                  // m for A frag, n for B frag
    const int khalf = lane >> 4;                  // 0 or 1

    const _Float16* arow[MSUB];
#pragma unroll
    for (int ms = 0; ms < MSUB; ++ms)
        arow[ms] = A + (size_t)(mg * (16 * MSUB) + ms * 16 + mn) * K;

    const _Float16* wrow[NSUB];
    int  ncol[NSUB];
    bool nval[NSUB];
#pragma unroll
    for (int s = 0; s < NSUB; ++s) {
        const int n = ng * (16 * NSUB) + s * 16 + mn;
        nval[s] = (n < N);
        ncol[s] = n;
        wrow[s] = W + (size_t)(nval[s] ? n : 0) * K;   // clamp: no OOB reads
    }

    v8f acc[MSUB][NSUB];
#pragma unroll
    for (int ms = 0; ms < MSUB; ++ms)
#pragma unroll
        for (int s = 0; s < NSUB; ++s) acc[ms][s] = (v8f){};

    for (int k0 = 0; k0 < K; k0 += 32) {
        const int kb = k0 + khalf * 8;
        __builtin_prefetch(arow[0] + k0 + 256, 0, 1);   // global_prefetch_b8
        v16h a[MSUB];
#pragma unroll
        for (int ms = 0; ms < MSUB; ++ms)
            a[ms] = load_frag_h(arow[ms] + kb);
#pragma unroll
        for (int s = 0; s < NSUB; ++s) {
            const v16h b = load_frag_h(wrow[s] + kb);
#pragma unroll
            for (int ms = 0; ms < MSUB; ++ms) {
                acc[ms][s] = __builtin_amdgcn_wmma_f32_16x16x32_f16(
                    /*neg_a=*/false, a[ms], /*neg_b=*/false, b,
                    /*c_mod=*/(short)0, acc[ms][s],
                    /*reuse_a=*/false, /*reuse_b=*/false);
            }
        }
    }

    // D layout: lane = n + 16*(m/8); vgpr r -> m = mbase + r
#pragma unroll
    for (int ms = 0; ms < MSUB; ++ms) {
        const int mbase = mg * (16 * MSUB) + ms * 16 + 8 * khalf;
#pragma unroll
        for (int s = 0; s < NSUB; ++s) {
            if (nval[s]) {
#pragma unroll
                for (int r = 0; r < 8; ++r)
                    C[(size_t)(mbase + r) * N + ncol[s]] = acc[ms][s][r];
            }
        }
    }
}

// =====================================================================
// Conv (causal depthwise, window 4) + SiLU, and dt softplus / dA prep.
// Fully parallel across (b, t, c) since conv_state[t] = xBC[t-3..t].
// =====================================================================
__device__ __forceinline__ float silu_f(float v) {
    return v / (1.0f + __expf(-v));
}

__global__ __launch_bounds__(256)
void conv_dt_kernel(const float* __restrict__ zxbcdt,
                    const float* __restrict__ conv_w,
                    const float* __restrict__ conv_b,
                    const float* __restrict__ dt_bias,
                    const float* __restrict__ A_log,
                    float* __restrict__ xbc_act,
                    float* __restrict__ dtbuf,
                    float* __restrict__ dAbuf) {
    const int idx = blockIdx.x * blockDim.x + threadIdx.x;

    if (idx < NTOK * CONV_DIM) {
        const int c  = idx % CONV_DIM;
        const int bt = idx / CONV_DIM;
        const int t  = bt % SEQLEN;
        const int b  = bt / SEQLEN;
        float acc = conv_b[c];
#pragma unroll
        for (int k = 0; k < 4; ++k) {
            const int tt = t - 3 + k;              // conv_state slot k = time t-3+k
            if (tt >= 0)
                acc += zxbcdt[((size_t)(b * SEQLEN + tt)) * D_IN_PROJ + D_INNER + c]
                       * conv_w[c * 4 + k];
        }
        xbc_act[(size_t)bt * CONV_DIM + c] = silu_f(acc);
    }

    if (idx < NTOK * NHEADS) {
        const int h  = idx % NHEADS;
        const int bt = idx / NHEADS;
        const float raw = zxbcdt[(size_t)bt * D_IN_PROJ + D_INNER + CONV_DIM + h]
                          + dt_bias[h];
        const float dt = (raw > 20.0f) ? raw : logf(1.0f + __expf(raw));
        const float Ah = -__expf(A_log[h]);
        dtbuf[bt * NHEADS + h] = dt;
        dAbuf[bt * NHEADS + h] = __expf(dt * Ah);
    }
}

// =====================================================================
// Sequential SSM scan. One block per (b, h): state 64x128 fp32 lives in
// registers (32 per thread, 256 threads). Per step, B/C (128+128 f32)
// are staged through LDS; y is reduced across the 4 n-chunks via LDS.
// Output written directly in f16 for the out-projection WMMA GEMM.
// =====================================================================
__global__ __launch_bounds__(256)
void scan_kernel(const float* __restrict__ zxbcdt,
                 const float* __restrict__ xbc_act,
                 const float* __restrict__ dtbuf,
                 const float* __restrict__ dAbuf,
                 const float* __restrict__ D_param,
                 _Float16* __restrict__ yMatH) {
    const int bh = blockIdx.x;
    const int b  = bh / NHEADS;
    const int h  = bh % NHEADS;
    const int tid = threadIdx.x;
    const int p      = tid & 63;        // headdim row
    const int nchunk = tid >> 6;        // 0..3
    const int n0     = nchunk * 32;

    __shared__ float lB[D_STATE];
    __shared__ float lC[D_STATE];
    __shared__ float yred[4][HEADDIM];

    float s[32];
#pragma unroll
    for (int i = 0; i < 32; ++i) s[i] = 0.0f;

    const float Dh = D_param[h];

    for (int t = 0; t < SEQLEN; ++t) {
        const size_t bt = (size_t)b * SEQLEN + t;
        const float* __restrict__ xbc = xbc_act + bt * CONV_DIM;

        if (tid < 128) lB[tid]       = xbc[D_INNER + tid];
        else           lC[tid - 128] = xbc[D_INNER + D_STATE + (tid - 128)];

        const float x  = xbc[h * HEADDIM + p];
        const float dt = dtbuf[bt * NHEADS + h];
        const float dA = dAbuf[bt * NHEADS + h];
        __syncthreads();

        const float dtx = dt * x;
        float yp = 0.0f;
#pragma unroll
        for (int i = 0; i < 32; ++i) {
            const float sv = s[i] * dA + dtx * lB[n0 + i];
            s[i] = sv;
            yp += sv * lC[n0 + i];
        }
        yred[nchunk][p] = yp;
        __syncthreads();

        if (tid < HEADDIM) {
            const float y = yred[0][p] + yred[1][p] + yred[2][p] + yred[3][p]
                            + Dh * x;
            const float z = zxbcdt[bt * D_IN_PROJ + h * HEADDIM + p];
            yMatH[bt * D_INNER + h * HEADDIM + p] = (_Float16)(y * silu_f(z));
        }
        // next iteration's first __syncthreads orders lB/lC rewrite vs reads
    }
}

// =====================================================================
// Launch
// =====================================================================
extern "C" void kernel_launch(void* const* d_in, const int* in_sizes, int n_in,
                              void* d_out, int out_size, void* d_ws, size_t ws_size,
                              hipStream_t stream) {
    const float* u       = (const float*)d_in[0];
    const float* W_in    = (const float*)d_in[1];
    const float* conv_w  = (const float*)d_in[2];
    const float* conv_b  = (const float*)d_in[3];
    const float* dt_bias = (const float*)d_in[4];
    const float* A_log   = (const float*)d_in[5];
    const float* D_param = (const float*)d_in[6];
    const float* W_out   = (const float*)d_in[7];
    float* out = (float*)d_out;

    // workspace layout
    char* ws = (char*)d_ws;
    float* zxbcdt   = (float*)ws;                   ws += (size_t)NTOK * D_IN_PROJ * 4;
    float* xbc_act  = (float*)ws;                   ws += (size_t)NTOK * CONV_DIM * 4;
    float* dtbuf    = (float*)ws;                   ws += (size_t)NTOK * NHEADS * 4;
    float* dAbuf    = (float*)ws;                   ws += (size_t)NTOK * NHEADS * 4;
    _Float16* uH    = (_Float16*)ws;                ws += (size_t)NTOK * D_MODEL * 2;
    _Float16* WinH  = (_Float16*)ws;                ws += (size_t)D_IN_PROJ * D_MODEL * 2;
    _Float16* WoutH = (_Float16*)ws;                ws += (size_t)D_MODEL * D_INNER * 2;
    _Float16* yMatH = (_Float16*)ws;                ws += (size_t)NTOK * D_INNER * 2;

    // 0) one-shot fp32 -> f16 conversions of GEMM operands
    {
        const int nu = NTOK * D_MODEL / 4;
        f32_to_f16_kernel<<<(nu + 255) / 256, 256, 0, stream>>>(u, uH, nu);
        const int nw = D_IN_PROJ * D_MODEL / 4;
        f32_to_f16_kernel<<<(nw + 255) / 256, 256, 0, stream>>>(W_in, WinH, nw);
        const int no = D_MODEL * D_INNER / 4;
        f32_to_f16_kernel<<<(no + 255) / 256, 256, 0, stream>>>(W_out, WoutH, no);
    }

    // 1) in-projection GEMM for all timesteps: zxbcdt = u @ W_in^T
    {
        const int mgroups = NTOK / 32;                       // 32
        const int ngroups = (D_IN_PROJ + 63) / 64;           // 53
        const int tiles   = mgroups * ngroups;
        wmma_gemm_nt<<<(tiles + 3) / 4, 128, 0, stream>>>(uH, WinH, zxbcdt,
                                                          NTOK, D_IN_PROJ, D_MODEL);
    }

    // 2) causal conv + SiLU + dt/dA preparation (parallel over b,t)
    {
        const int total = NTOK * CONV_DIM;
        conv_dt_kernel<<<(total + 255) / 256, 256, 0, stream>>>(
            zxbcdt, conv_w, conv_b, dt_bias, A_log, xbc_act, dtbuf, dAbuf);
    }

    // 3) sequential SSM scan: 48 independent (b,h) recurrences
    scan_kernel<<<BATCH * NHEADS, 256, 0, stream>>>(
        zxbcdt, xbc_act, dtbuf, dAbuf, D_param, yMatH);

    // 4) out-projection GEMM: out = yMat @ W_out^T
    {
        const int mgroups = NTOK / 32;                       // 32
        const int ngroups = D_MODEL / 64;                    // 12
        const int tiles   = mgroups * ngroups;
        wmma_gemm_nt<<<(tiles + 3) / 4, 128, 0, stream>>>(yMatH, WoutH, out,
                                                          NTOK, D_MODEL, D_INNER);
    }
}
